// CrossAttention_90598040142135
// MI455X (gfx1250) — matrix operand order repaired and checked by reading, not compiled, not run on hardware
//
#include <hip/hip_runtime.h>
#include <hip/hip_bf16.h>

typedef __attribute__((ext_vector_type(16))) _Float16 v16h;
typedef __attribute__((ext_vector_type(8)))  _Float16 v8h;
typedef __attribute__((ext_vector_type(4)))  _Float16 v4h;
typedef __attribute__((ext_vector_type(8)))  float    v8f;

#define DQ 512
#define DC 768
#define NB 8
#define LQ 2048
#define LKV 2048

static __device__ __forceinline__ v8f wmma_f16(v16h a, v16h b, v8f c) {
    return __builtin_amdgcn_wmma_f32_16x16x32_f16(false, a, false, b, (short)0, c, false, false);
}

// ---------------------------------------------------------------------------
// Prep: W [K][N] f32  ->  WT [N][K] f16   (B-matrix layout: K contiguous)
// ---------------------------------------------------------------------------
__global__ void transpose_w_kernel(const float* __restrict__ W, _Float16* __restrict__ WT,
                                   int K, int N) {
    int idx = blockIdx.x * blockDim.x + threadIdx.x;
    if (idx >= K * N) return;
    int n = idx / K;
    int k = idx - n * K;
    WT[idx] = (_Float16)W[k * N + n];
}

// ---------------------------------------------------------------------------
// WMMA GEMM: C[row, n] = A[row, 0:K] * BT[n, 0:K] + bias[n]
// One wave computes a 16-row x 64-col tile; A/B groups double-buffered
// across the k-loop (8 loads in flight during the WMMAs).
// AF16: 0 = A is f32, 1 = A is f16
// OUTMODE: 0 = f16 row-major; 1 = f16 per-batch transposed (VprojT); 2 = f32
// ---------------------------------------------------------------------------
template <int AF16>
static __device__ __forceinline__ v16h load_a_frag(const void* Aptr, int row, int K, int kb) {
    v16h a;
    if (AF16) {
        const _Float16* Arow = (const _Float16*)Aptr + (size_t)row * K;
        v8h a0 = *(const v8h*)(Arow + kb);
        v8h a1 = *(const v8h*)(Arow + kb + 8);
#pragma unroll
        for (int e = 0; e < 8; ++e) { a[e] = a0[e]; a[8 + e] = a1[e]; }
    } else {
        const float* Arow = (const float*)Aptr + (size_t)row * K;
        float4 x0 = *(const float4*)(Arow + kb);
        float4 x1 = *(const float4*)(Arow + kb + 4);
        float4 y0 = *(const float4*)(Arow + kb + 8);
        float4 y1 = *(const float4*)(Arow + kb + 12);
        a[0]=(_Float16)x0.x; a[1]=(_Float16)x0.y; a[2]=(_Float16)x0.z; a[3]=(_Float16)x0.w;
        a[4]=(_Float16)x1.x; a[5]=(_Float16)x1.y; a[6]=(_Float16)x1.z; a[7]=(_Float16)x1.w;
        a[8]=(_Float16)y0.x; a[9]=(_Float16)y0.y; a[10]=(_Float16)y0.z; a[11]=(_Float16)y0.w;
        a[12]=(_Float16)y1.x; a[13]=(_Float16)y1.y; a[14]=(_Float16)y1.z; a[15]=(_Float16)y1.w;
    }
    return a;
}

template <int AF16, int OUTMODE>
__global__ __launch_bounds__(32)
void gemm_wmma_kernel(const void* __restrict__ Aptr,
                      const _Float16* __restrict__ BT,
                      const float* __restrict__ bias,
                      void* __restrict__ Cptr,
                      int K, int N) {
    const int lane  = threadIdx.x;
    const int lrow  = lane & 15;
    const int lhalf = lane >> 4;
    const int rowbase = blockIdx.x * 16;
    const int nbase   = blockIdx.y * 64;

    v8f acc[4] = {};

    const int nchunks = K >> 5;  // K/32
    v16h a = load_a_frag<AF16>(Aptr, rowbase + lrow, K, lhalf * 16);
    v16h bfr[4];
#pragma unroll
    for (int t = 0; t < 4; ++t)
        bfr[t] = *(const v16h*)(BT + (size_t)(nbase + t * 16 + lrow) * K + lhalf * 16);

    for (int kc = 0; kc < nchunks; ++kc) {
        // ---- issue next chunk's A + B loads before consuming current
        v16h a_next = a;
        v16h bnx[4];
#pragma unroll
        for (int t = 0; t < 4; ++t) bnx[t] = bfr[t];
        if (kc + 1 < nchunks) {
            const int ko = ((kc + 1) << 5) + lhalf * 16;
#pragma unroll
            for (int t = 0; t < 4; ++t)
                bnx[t] = *(const v16h*)(BT + (size_t)(nbase + t * 16 + lrow) * K + ko);
            a_next = load_a_frag<AF16>(Aptr, rowbase + lrow, K, ((kc + 1) << 5) + lhalf * 16);
        }
#pragma unroll
        for (int t = 0; t < 4; ++t) acc[t] = wmma_f16(a, bfr[t], acc[t]);
        a = a_next;
#pragma unroll
        for (int t = 0; t < 4; ++t) bfr[t] = bnx[t];
    }

    // ---- epilogue: C tile layout: VGPR r -> (M = r + 8*half, N = lane&15)
#pragma unroll
    for (int t = 0; t < 4; ++t) {
        const int n = nbase + t * 16 + lrow;
        const float bval = bias[n];
        if (OUTMODE == 1) {
            const int bb = rowbase >> 11;
            const int kvb = (rowbase & (LKV - 1)) + lhalf * 8;
            v8h packed;
#pragma unroll
            for (int r = 0; r < 8; ++r) packed[r] = (_Float16)(acc[t][r] + bval);
            *(v8h*)((_Float16*)Cptr + ((size_t)bb * N + n) * LKV + kvb) = packed;
        } else {
#pragma unroll
            for (int r = 0; r < 8; ++r) {
                const int Mg = rowbase + r + 8 * lhalf;
                const float v = acc[t][r] + bval;
                if (OUTMODE == 0) ((_Float16*)Cptr)[(size_t)Mg * N + n] = (_Float16)v;
                else              ((float*)Cptr)[(size_t)Mg * N + n] = v;
            }
        }
    }
}

// ---------------------------------------------------------------------------
// Flash attention: 2 waves cooperate on one 16-row Q tile.
//   wave w: scores N-tile kv [16w,16w+16);  PV d-slice [256w, 256w+256)
// K/V fragment streams are software-pipelined at distance 3 (6 b128 loads
// in flight during each WMMA).
// ---------------------------------------------------------------------------
__global__ __launch_bounds__(64)
void flash_attn_kernel(const float* __restrict__ query,
                       const _Float16* __restrict__ Kproj,
                       const _Float16* __restrict__ VprojT,
                       _Float16* __restrict__ ctx) {
    __shared__ __align__(16) _Float16 Qlds[16 * DQ];     // 16 KB
    __shared__ __align__(16) _Float16 Plds[2][16 * 32];  // 2 KB double-buffered
    __shared__ float RedMax[2][2][16];
    __shared__ float RedSum[2][2][16];

    const int tid   = threadIdx.x;
    const int wv    = tid >> 5;   // wave id 0/1
    const int lane  = tid & 31;
    const int lrow  = lane & 15;
    const int lhalf = lane >> 4;
    const int b  = blockIdx.y;
    const int q0 = blockIdx.x * 16;
    const float scale = 0.044194173824159216f;  // 1/sqrt(512)

    // ---- cooperative Q load into LDS (f32 -> f16), 4 threads per row
    {
        const int row = tid >> 2;
        const int c0  = (tid & 3) * 128;
        const float* src = query + ((size_t)b * LQ + q0 + row) * DQ + c0;
        _Float16* dst = &Qlds[row * DQ + c0];
#pragma unroll 4
        for (int i = 0; i < 128; i += 4) {
            float4 x = *(const float4*)(src + i);
            v4h h; h[0]=(_Float16)x.x; h[1]=(_Float16)x.y; h[2]=(_Float16)x.z; h[3]=(_Float16)x.w;
            *(v4h*)(dst + i) = h;
        }
    }
    __syncthreads();

    v8f o[16];
#pragma unroll
    for (int t = 0; t < 16; ++t) o[t] = (v8f)0.f;
    float m_r[8], l_r[8];
#pragma unroll
    for (int r = 0; r < 8; ++r) { m_r[r] = -1e30f; l_r[r] = 0.f; }

    const _Float16* Kb = Kproj  + (size_t)b * LKV * DQ;
    const _Float16* Vb = VprojT + (size_t)b * DQ * LKV;
    const _Float16* qbase = &Qlds[lrow * DQ + lhalf * 16];

#pragma unroll 1
    for (int kvb = 0; kvb < LKV / 32; ++kvb) {
        const int kv0 = kvb * 32;
        const int buf = kvb & 1;

        const _Float16* Krow = Kb + (size_t)(kv0 + wv * 16 + lrow) * DQ + lhalf * 16;
        const _Float16* VrowBase = Vb + (size_t)(wv * 256 + lrow) * LKV + kv0 + lhalf * 16;

        // ---- L2 warm-up for the block 2 iterations ahead
        if (kvb + 2 < LKV / 32) {
            __builtin_prefetch((const void*)(Krow + 64 * DQ), 0, 1);
            __builtin_prefetch((const void*)(VrowBase + 64), 0, 1);
        }

        // ---- scores for this wave's 16-col kv tile: S = Q * Kproj^T
        //      (K fragments pipelined at distance 3)
        v8f s = (v8f)0.f;
        v16h kf0 = *(const v16h*)(Krow);
        v16h kf1 = *(const v16h*)(Krow + 32);
        v16h kf2 = *(const v16h*)(Krow + 64);
#pragma unroll
        for (int c = 0; c < 16; ++c) {
            v16h knew = kf0;
            if (c + 3 < 16) knew = *(const v16h*)(Krow + (c + 3) * 32);
            v8h q0v = *(const v8h*)(qbase + c * 32);
            v8h q1v = *(const v8h*)(qbase + c * 32 + 8);
            v16h qf;
#pragma unroll
            for (int e = 0; e < 8; ++e) { qf[e] = q0v[e]; qf[8 + e] = q1v[e]; }
            s = wmma_f16(qf, kf0, s);
            kf0 = kf1; kf1 = kf2; kf2 = knew;
        }

        // ---- phase 1: partial row max over own 16 columns
        float mx8[8];
#pragma unroll
        for (int r = 0; r < 8; ++r) {
            float sv = s[r] * scale;
            s[r] = sv;
            float mx = sv;
            mx = fmaxf(mx, __shfl_xor(mx, 8, 16));
            mx = fmaxf(mx, __shfl_xor(mx, 4, 16));
            mx = fmaxf(mx, __shfl_xor(mx, 2, 16));
            mx = fmaxf(mx, __shfl_xor(mx, 1, 16));
            mx8[r] = mx;
            if (lrow == r) RedMax[buf][wv][r + 8 * lhalf] = mx;
        }
        __syncthreads();

        // ---- phase 2: combine maxes, exponentiate, rescale O, publish P + partial sums
        float alpha8[8], rsown8[8];
#pragma unroll
        for (int r = 0; r < 8; ++r) {
            const float mo    = RedMax[buf][wv ^ 1][r + 8 * lhalf];
            const float mnew  = fmaxf(m_r[r], fmaxf(mx8[r], mo));
            const float alpha = __expf(m_r[r] - mnew);
            m_r[r] = mnew;
            alpha8[r] = alpha;
            const float p = __expf(s[r] - mnew);
            float rs = p;
            rs += __shfl_xor(rs, 8, 16);
            rs += __shfl_xor(rs, 4, 16);
            rs += __shfl_xor(rs, 2, 16);
            rs += __shfl_xor(rs, 1, 16);
            rsown8[r] = rs;
            if (lrow == r) RedSum[buf][wv][r + 8 * lhalf] = rs;
            Plds[buf][(r + 8 * lhalf) * 32 + wv * 16 + lrow] = (_Float16)p;
#pragma unroll
            for (int t = 0; t < 16; ++t) o[t][r] *= alpha;
        }
        __syncthreads();

        // ---- phase 3: finish l, read full P as A-fragment, PV on own d-slice
#pragma unroll
        for (int r = 0; r < 8; ++r) {
            const float ro = RedSum[buf][wv ^ 1][r + 8 * lhalf];
            l_r[r] = l_r[r] * alpha8[r] + rsown8[r] + ro;
        }
        v16h pf;
        {
            const _Float16* prow = &Plds[buf][lrow * 32 + lhalf * 16];
            v8h p0v = *(const v8h*)(prow);
            v8h p1v = *(const v8h*)(prow + 8);
#pragma unroll
            for (int e = 0; e < 8; ++e) { pf[e] = p0v[e]; pf[8 + e] = p1v[e]; }
        }
        // ---- PV, V fragments pipelined at distance 3
        v16h vf0 = *(const v16h*)(VrowBase);
        v16h vf1 = *(const v16h*)(VrowBase + (size_t)16 * LKV);
        v16h vf2 = *(const v16h*)(VrowBase + (size_t)32 * LKV);
#pragma unroll
        for (int t = 0; t < 16; ++t) {
            v16h vnew = vf0;
            if (t + 3 < 16) vnew = *(const v16h*)(VrowBase + (size_t)(t + 3) * 16 * LKV);
            o[t] = wmma_f16(pf, vf0, o[t]);
            vf0 = vf1; vf1 = vf2; vf2 = vnew;
        }
    }

    // ---- normalize and store this wave's 256-wide d-slice of the context
#pragma unroll
    for (int r = 0; r < 8; ++r) {
        const float inv = 1.f / l_r[r];
        const size_t Mg = (size_t)b * LQ + q0 + r + 8 * lhalf;
#pragma unroll
        for (int t = 0; t < 16; ++t)
            ctx[Mg * DQ + wv * 256 + t * 16 + lrow] = (_Float16)(o[t][r] * inv);
    }
}

// ---------------------------------------------------------------------------
extern "C" void kernel_launch(void* const* d_in, const int* in_sizes, int n_in,
                              void* d_out, int out_size, void* d_ws, size_t ws_size,
                              hipStream_t stream) {
    const float* query = (const float*)d_in[0];
    const float* key   = (const float*)d_in[1];
    const float* value = (const float*)d_in[2];
    const float* Wk    = (const float*)d_in[3];
    const float* bk    = (const float*)d_in[4];
    const float* Wv    = (const float*)d_in[5];
    const float* bv    = (const float*)d_in[6];
    const float* Wo    = (const float*)d_in[7];
    const float* bo    = (const float*)d_in[8];
    float* out = (float*)d_out;

    _Float16* ws  = (_Float16*)d_ws;
    _Float16* WkT = ws;                                   // 512*768
    _Float16* WvT = WkT + (size_t)DQ * DC;                // 512*768
    _Float16* WoT = WvT + (size_t)DQ * DC;                // 512*512
    _Float16* Kp  = WoT + (size_t)DQ * DQ;                // 8*2048*512
    _Float16* VpT = Kp  + (size_t)NB * LKV * DQ;          // 8*2048*512
    _Float16* Ctx = VpT + (size_t)NB * LKV * DQ;          // 8*2048*512

    const int rows = NB * LKV;  // 16384

    transpose_w_kernel<<<(DC * DQ + 255) / 256, 256, 0, stream>>>(Wk, WkT, DC, DQ);
    transpose_w_kernel<<<(DC * DQ + 255) / 256, 256, 0, stream>>>(Wv, WvT, DC, DQ);
    transpose_w_kernel<<<(DQ * DQ + 255) / 256, 256, 0, stream>>>(Wo, WoT, DQ, DQ);

    gemm_wmma_kernel<0, 0><<<dim3(rows / 16, DQ / 64), 32, 0, stream>>>(
        (const void*)key, WkT, bk, (void*)Kp, DC, DQ);
    gemm_wmma_kernel<0, 1><<<dim3(rows / 16, DQ / 64), 32, 0, stream>>>(
        (const void*)value, WvT, bv, (void*)VpT, DC, DQ);

    flash_attn_kernel<<<dim3(LQ / 16, NB), 64, 0, stream>>>(query, Kp, VpT, Ctx);

    gemm_wmma_kernel<1, 2><<<dim3(rows / 16, DQ / 64), 32, 0, stream>>>(
        (const void*)Ctx, WoT, bo, (void*)out, DQ, DQ);
}
